// _Mamba_78159814852916
// MI455X (gfx1250) — matrix operand order repaired and checked by reading, not compiled, not run on hardware
//
#include <hip/hip_runtime.h>
#include <hip/hip_bf16.h>

// ---------------------------------------------------------------------------
// Mamba forward for MI455X (gfx1250, wave32).
// GEMMs in bf16 via v_wmma_f32_16x16x32_bf16 (fp32 accumulate); conv/scan fp32.
// GEMM: block tile 128x128, wave tile 32x64 (2x4 WMMA frags), double-buffered
// LDS fed by GLOBAL_LOAD_ASYNC_TO_LDS_B128 (ASYNCcnt), one barrier per K-step.
// ---------------------------------------------------------------------------

typedef __attribute__((ext_vector_type(16))) __bf16 v16bf;
typedef __attribute__((ext_vector_type(8)))  __bf16 v8bf;
typedef __attribute__((ext_vector_type(8)))  float  v8f;

#define D_MODEL  1024
#define D_STATE  16
#define D_CONV   4
#define D_INNER  2048
#define DT_RANK  64
#define BATCH    2
#define SEQLEN   1024
#define MROWS    (BATCH * SEQLEN)      // 2048 token rows

__device__ __forceinline__ unsigned short f32_to_bf16(float f) {
  unsigned u = __float_as_uint(f);
  unsigned r = u + 0x7fffu + ((u >> 16) & 1u);   // round-to-nearest-even
  if ((u & 0x7f800000u) == 0x7f800000u) r = u;   // keep inf/nan payload
  return (unsigned short)(r >> 16);
}

// LDS byte offset (AS(3) address) of a __shared__ pointer.
typedef __attribute__((address_space(3))) unsigned short lds_u16;
__device__ __forceinline__ unsigned lds_off_of(const unsigned short* p) {
  return (unsigned)(unsigned long long)(lds_u16*)(void*)p;
}

// Async global->LDS copy of 32B (two B128 beats); INST_OFFSET is added to both
// the global and the LDS address (ISA 08_async_tensor 4.4), so one address
// pair covers both beats. Tracked by ASYNCcnt.
__device__ __forceinline__ void async_copy32(unsigned lds_off, const void* gptr) {
  asm volatile(
      "global_load_async_to_lds_b128 %0, %1, off\n\t"
      "global_load_async_to_lds_b128 %0, %1, off offset:16"
      :: "v"(lds_off), "v"(gptr) : "memory");
}
__device__ __forceinline__ void wait_async0() {
  asm volatile("s_wait_asynccnt 0x0" ::: "memory");
}

// -------------------------------- fp32 -> bf16 ------------------------------
__global__ void cvt_f32_bf16(const float* __restrict__ src,
                             unsigned short* __restrict__ dst, int n) {
  int i = blockIdx.x * 256 + threadIdx.x;
  if (i < n) dst[i] = f32_to_bf16(src[i]);
}

// ------------------------- bf16 WMMA GEMM:  C = A * W^T ---------------------
// A: [M x K] bf16 row-major (tokens), W: [N x K] bf16 row-major (weights),
// C: [M x N] fp32. M % 128 == 0, K % 32 == 0; N guarded.
// Block: 256 threads (8 waves), tile 128x128. Wave: 32x64 (2x4 WMMA frags).
#define LDT 40   // LDS row stride in bf16 elements (80B: 16B-aligned, bank-spread)

__global__ __launch_bounds__(256) void gemm_bf16_nt(
    const unsigned short* __restrict__ A,
    const unsigned short* __restrict__ W,
    float* __restrict__ C,
    int M, int N, int K)
{
  __shared__ __align__(16) unsigned short As[2][128 * LDT];  // 2 x 10 KB
  __shared__ __align__(16) unsigned short Bs[2][128 * LDT];  // 2 x 10 KB

  const int tid    = threadIdx.x;
  const int wave   = tid >> 5;
  const int lane   = tid & 31;
  const int laneLo = lane & 15;
  const int laneHi = lane >> 4;
  const int waveM  = wave & 3;         // 0..3  -> 32-row strip
  const int waveN  = wave >> 2;        // 0..1  -> 64-col strip
  const int gM0    = blockIdx.y * 128;
  const int gN0    = blockIdx.x * 128;

  // staging: each thread owns one 16-elem row-segment of A and B per K-step
  const int sRow = tid >> 1;           // 0..127
  const int sSeg = (tid & 1) * 16;     // 0 / 16
  int bRowG = gN0 + sRow; if (bRowG > N - 1) bRowG = N - 1;  // clamp (N=96 case)

  const unsigned short* gA = A + (size_t)(gM0 + sRow) * K + sSeg;
  const unsigned short* gB = W + (size_t)bRowG        * K + sSeg;

  unsigned ldsA[2], ldsB[2];
  ldsA[0] = lds_off_of(&As[0][sRow * LDT + sSeg]);
  ldsA[1] = lds_off_of(&As[1][sRow * LDT + sSeg]);
  ldsB[0] = lds_off_of(&Bs[0][sRow * LDT + sSeg]);
  ldsB[1] = lds_off_of(&Bs[1][sRow * LDT + sSeg]);

  union Frag { v16bf f; v8bf h[2]; };

  v8f acc[2][4];
  #pragma unroll
  for (int i = 0; i < 2; ++i)
    #pragma unroll
    for (int j = 0; j < 4; ++j)
      #pragma unroll
      for (int r = 0; r < 8; ++r) acc[i][j][r] = 0.0f;

  // ---- prologue: async-stage K-tile 0 into buffer 0 ----
  async_copy32(ldsA[0], gA);
  async_copy32(ldsB[0], gB);
  wait_async0();
  __syncthreads();

  const int nIter = K >> 5;            // K / 32
  for (int it = 0; it < nIter; ++it) {
    const int cur = it & 1;
    const bool more = (it + 1) < nIter;

    // fire-and-forget async stage of the next K-tile (hides under WMMAs)
    if (more) {
      async_copy32(ldsA[cur ^ 1], gA + (size_t)(it + 1) * 32);
      async_copy32(ldsB[cur ^ 1], gB + (size_t)(it + 1) * 32);
    }

    // ---- fragment gathers (ISA 16-bit A 16x32 / B 32x16 lane layouts) ----
    const unsigned short* as = As[cur];
    const unsigned short* bs = Bs[cur];
    Frag a[2], b[4];
    #pragma unroll
    for (int t = 0; t < 2; ++t) {
      int m = waveM * 32 + t * 16 + laneLo;
      a[t].h[0] = *(const v8bf*)&as[m * LDT + 8 * laneHi];       // K = 8*hi..+7
      a[t].h[1] = *(const v8bf*)&as[m * LDT + 16 + 8 * laneHi];  // K = 16+8*hi..+7
    }
    #pragma unroll
    for (int t = 0; t < 4; ++t) {
      int n = waveN * 64 + t * 16 + laneLo;
      b[t].h[0] = *(const v8bf*)&bs[n * LDT + 8 * laneHi];       // K = 8*hi..+7
      b[t].h[1] = *(const v8bf*)&bs[n * LDT + 16 + 8 * laneHi];  // K = 16+8*hi..+7
    }

    #pragma unroll
    for (int i = 0; i < 2; ++i)
      #pragma unroll
      for (int j = 0; j < 4; ++j)
        acc[i][j] = __builtin_amdgcn_wmma_f32_16x16x32_bf16(
            false, a[i].f, false, b[j].f, (short)0, acc[i][j], false, false);

    // async writes must land + be visible before anyone reads buf[nxt]
    if (more) {
      wait_async0();
      __syncthreads();
    }
  }

  // ---- store (C/D layout: elem r -> row r + 8*laneHi, col laneLo) ----
  #pragma unroll
  for (int i = 0; i < 2; ++i) {
    #pragma unroll
    for (int j = 0; j < 4; ++j) {
      int col = gN0 + waveN * 64 + j * 16 + laneLo;
      if (col < N) {
        #pragma unroll
        for (int r = 0; r < 8; ++r) {
          int row = gM0 + waveM * 32 + i * 16 + 8 * laneHi + r;
          C[(size_t)row * N + col] = acc[i][j][r];
        }
      }
    }
  }
}

// -------------------- depthwise causal conv (k=4) + SiLU --------------------
// xz: [MROWS x 4096] fp32 (x = cols 0..2047). Emits u fp32 and u bf16.
__global__ void conv_silu(const float* __restrict__ xz,
                          const float* __restrict__ w,      // [D_INNER x 4]
                          const float* __restrict__ bias,   // [D_INNER]
                          float* __restrict__ u,
                          unsigned short* __restrict__ u16) {
  int idx = blockIdx.x * 256 + threadIdx.x;
  if (idx >= MROWS * D_INNER) return;
  int d   = idx & (D_INNER - 1);
  int row = idx >> 11;                // / 2048
  int l   = row & (SEQLEN - 1);
  int b   = row >> 10;
  float acc = bias[d];
  #pragma unroll
  for (int j = 0; j < D_CONV; ++j) {
    int li = l - (D_CONV - 1) + j;
    if (li >= 0)
      acc += xz[(size_t)(b * SEQLEN + li) * (2 * D_INNER) + d] * w[d * D_CONV + j];
  }
  float s = acc / (1.0f + __expf(-acc));   // SiLU
  u[idx]   = s;
  u16[idx] = f32_to_bf16(s);
}

// ------------------- slice dt_low (cols 0..63 of x_dbl) to bf16 -------------
__global__ void extract_dtlow(const float* __restrict__ xdbl,
                              unsigned short* __restrict__ dt16) {
  int idx = blockIdx.x * 256 + threadIdx.x;      // MROWS * 64
  if (idx >= MROWS * DT_RANK) return;
  int r = idx >> 6, c = idx & 63;
  dt16[idx] = f32_to_bf16(xdbl[r * 96 + c]);
}

// --------------------------- selective scan ---------------------------------
// One thread per (b, d) channel; 16-state recurrence in registers.
__global__ __launch_bounds__(256) void selective_scan(
    const float* __restrict__ delta_raw,   // [MROWS x D_INNER] (pre-bias)
    const float* __restrict__ dt_bias,     // [D_INNER]
    const float* __restrict__ u,           // [MROWS x D_INNER]
    const float* __restrict__ xdbl,        // [MROWS x 96] (B: 64..79, C: 80..95)
    const float* __restrict__ A_log,       // [D_INNER x D_STATE]
    const float* __restrict__ Dp,          // [D_INNER]
    const float* __restrict__ xz,          // [MROWS x 4096] (z = cols 2048..)
    unsigned short* __restrict__ outz16)   // [MROWS x D_INNER]
{
  int b = blockIdx.x >> 3;                            // 16 blocks: 8 per batch
  int d = (blockIdx.x & 7) * 256 + threadIdx.x;

  float Anb[D_STATE], h[D_STATE];
  #pragma unroll
  for (int n = 0; n < D_STATE; ++n) {
    Anb[n] = -__expf(A_log[d * D_STATE + n]);
    h[n]   = 0.0f;
  }
  const float bias = dt_bias[d];
  const float Dv   = Dp[d];

  for (int l = 0; l < SEQLEN; ++l) {
    size_t row  = (size_t)b * SEQLEN + l;
    float draw  = delta_raw[row * D_INNER + d] + bias;
    float delta = (draw > 20.0f) ? draw : __logf(1.0f + __expf(draw));  // softplus
    float uv    = u[row * D_INNER + d];
    const float4* Bp = (const float4*)(xdbl + row * 96 + 64);
    const float4* Cp = (const float4*)(xdbl + row * 96 + 80);
    float y = 0.0f;
    #pragma unroll
    for (int q = 0; q < 4; ++q) {
      float4 Bv = Bp[q], Cv = Cp[q];
      float bb[4] = {Bv.x, Bv.y, Bv.z, Bv.w};
      float cc[4] = {Cv.x, Cv.y, Cv.z, Cv.w};
      #pragma unroll
      for (int e = 0; e < 4; ++e) {
        int n = q * 4 + e;
        float dA = __expf(delta * Anb[n]);
        h[n] = dA * h[n] + delta * bb[e] * uv;
        y   += h[n] * cc[e];
      }
    }
    float yb = y + uv * Dv;
    float z  = xz[row * (2 * D_INNER) + D_INNER + d];
    float oz = yb * (z / (1.0f + __expf(-z)));       // * silu(z)
    outz16[row * D_INNER + d] = f32_to_bf16(oz);
  }
}

// ---------------------------------------------------------------------------
extern "C" void kernel_launch(void* const* d_in, const int* in_sizes, int n_in,
                              void* d_out, int out_size, void* d_ws, size_t ws_size,
                              hipStream_t stream) {
  const float* hs        = (const float*)d_in[0];   // [2,1024,1024]
  const float* in_proj_w = (const float*)d_in[1];   // [4096,1024]
  const float* conv_w    = (const float*)d_in[2];   // [2048,4]
  const float* conv_b    = (const float*)d_in[3];   // [2048]
  const float* x_proj_w  = (const float*)d_in[4];   // [96,2048]
  const float* dt_proj_w = (const float*)d_in[5];   // [2048,64]
  const float* dt_proj_b = (const float*)d_in[6];   // [2048]
  const float* A_log     = (const float*)d_in[7];   // [2048,16]
  const float* D_param   = (const float*)d_in[8];   // [2048]
  const float* out_proj_w= (const float*)d_in[9];   // [1024,2048]
  float* out             = (float*)d_out;           // [2,1024,1024] fp32

  char* ws = (char*)d_ws;
  size_t off = 0;
  auto carve = [&](size_t bytes) -> void* {
    void* p = ws + off;
    off = (off + bytes + 255) & ~(size_t)255;
    return p;
  };

  unsigned short* hs16   = (unsigned short*)carve((size_t)MROWS * D_MODEL * 2);
  unsigned short* win16  = (unsigned short*)carve((size_t)2 * D_INNER * D_MODEL * 2);
  unsigned short* wout16 = (unsigned short*)carve((size_t)D_MODEL * D_INNER * 2);
  unsigned short* wxp16  = (unsigned short*)carve((size_t)96 * D_INNER * 2);
  unsigned short* wdt16  = (unsigned short*)carve((size_t)D_INNER * DT_RANK * 2);
  float*          xz     = (float*)carve((size_t)MROWS * 2 * D_INNER * 4);
  float*          u      = (float*)carve((size_t)MROWS * D_INNER * 4);
  unsigned short* u16    = (unsigned short*)carve((size_t)MROWS * D_INNER * 2);
  float*          xdbl   = (float*)carve((size_t)MROWS * 96 * 4);
  unsigned short* dt16   = (unsigned short*)carve((size_t)MROWS * DT_RANK * 2);
  float*          draw   = (float*)carve((size_t)MROWS * D_INNER * 4);
  unsigned short* oz16   = (unsigned short*)carve((size_t)MROWS * D_INNER * 2);
  (void)ws_size; (void)in_sizes; (void)n_in; (void)out_size;

  auto cvt = [&](const float* s, unsigned short* d, int n) {
    cvt_f32_bf16<<<(n + 255) / 256, 256, 0, stream>>>(s, d, n);
  };
  cvt(hs,         hs16,   MROWS * D_MODEL);
  cvt(in_proj_w,  win16,  2 * D_INNER * D_MODEL);
  cvt(out_proj_w, wout16, D_MODEL * D_INNER);
  cvt(x_proj_w,   wxp16,  96 * D_INNER);
  cvt(dt_proj_w,  wdt16,  D_INNER * DT_RANK);

  // 1) xz = hs @ in_proj_w^T          [2048 x 4096], K=1024
  gemm_bf16_nt<<<dim3(4096 / 128, MROWS / 128), 256, 0, stream>>>(
      hs16, win16, xz, MROWS, 2 * D_INNER, D_MODEL);

  // 2) u = silu(causal_conv(x) + b)   [2048 x 2048]
  conv_silu<<<(MROWS * D_INNER + 255) / 256, 256, 0, stream>>>(
      xz, conv_w, conv_b, u, u16);

  // 3) x_dbl = u @ x_proj_w^T         [2048 x 96], K=2048 (N guarded)
  gemm_bf16_nt<<<dim3(1, MROWS / 128), 256, 0, stream>>>(
      u16, wxp16, xdbl, MROWS, 96, D_INNER);

  // 4) delta_raw = dt_low @ dt_proj_w^T  [2048 x 2048], K=64
  extract_dtlow<<<(MROWS * DT_RANK + 255) / 256, 256, 0, stream>>>(xdbl, dt16);
  gemm_bf16_nt<<<dim3(D_INNER / 128, MROWS / 128), 256, 0, stream>>>(
      dt16, wdt16, draw, MROWS, D_INNER, DT_RANK);

  // 5) selective scan (+ D skip, * silu(z)), emits bf16 out_z
  selective_scan<<<16, 256, 0, stream>>>(
      draw, dt_proj_b, u, xdbl, A_log, D_param, xz, oz16);

  // 6) out = out_z @ out_proj_w^T     [2048 x 1024], K=2048 -> fp32 d_out
  gemm_bf16_nt<<<dim3(D_MODEL / 128, MROWS / 128), 256, 0, stream>>>(
      oz16, wout16, out, MROWS, D_MODEL, D_INNER);
}
